// TokenCompressor_23570780520557
// MI455X (gfx1250) — matrix operand order repaired and checked by reading, not compiled, not run on hardware
//
#include <hip/hip_runtime.h>
#include <hip/hip_bf16.h>
#include <math.h>

typedef _Float16 v8h  __attribute__((ext_vector_type(8)));
typedef _Float16 v16h __attribute__((ext_vector_type(16)));
typedef float    v8f  __attribute__((ext_vector_type(8)));

#define TC_B   8
#define TC_N   16384
#define TC_C   128
#define TC_NB  1023      // (16384-32)/16 + 1
#define TC_K1  4096      // 32*128
#define TC_N1  256
#define TC_N2  128
#define CHUNK  1024      // K-chunk staged in LDS (8 tokens x 128 ch)
#define ASTR   (CHUNK + 8)   // +16B row pad -> LDS banks spread
#define HSTR   (TC_N1 + 8)
#define MROWS  32        // block-rows per workgroup (2 x 16-row WMMA tiles)

// ---------------------------------------------------------------------------
// Pre-pass: swizzle fp32 row-major W (K x N) into f16 WMMA B-fragment order.
// B-matrix 16-bit layout (ISA 7.12.4): lanes 0-15 hold K=kt*32+{0..7,16..23} (col =
// lane%16), lanes 16-31 hold K=kt*32+{8..15,24..31}. One fragment = 32 contiguous
// bytes per lane at ((nt*ktiles + kt)*32 + lane)*16 halves.
// ---------------------------------------------------------------------------
__global__ __launch_bounds__(256)
void tc_swizzleW(const float* __restrict__ W, _Float16* __restrict__ out,
                 int ntiles, int ktiles, int N)
{
    int t = blockIdx.x * blockDim.x + threadIdx.x;
    int total = ntiles * ktiles * 32;
    if (t >= total) return;
    int lane = t & 31;
    int rem  = t >> 5;
    int kt   = rem % ktiles;
    int half = lane >> 4;
    int n    = ((rem / ktiles) << 4) + (lane & 15);
    v16h frag;
#pragma unroll
    for (int e = 0; e < 16; ++e) {
        int k = kt * 32 + (e < 8 ? half * 8 + e : 16 + half * 8 + (e - 8));
        frag[e] = (_Float16)W[(size_t)k * N + n];
    }
    *(v16h*)(out + (size_t)t * 16) = frag;
}

// ---------------------------------------------------------------------------
// Main fused kernel: window gather(+pos) -> GEMM1(f16 WMMA, f32 acc) ->
// bias + exact-erf GELU -> GEMM2 -> bias -> f32 store.
// Grid: 8 batches x 32 M-tiles (32 block-rows each). Block: 8 wave32s.
// Each wave: 32 rows x 32 cols of GEMM1 (4 accumulators); per k-step it
// loads 2 A-frags (LDS) + 2 B-frags (global, L2-resident) and issues 4 WMMA.
// ---------------------------------------------------------------------------
__global__ __launch_bounds__(256)
void tc_main(const float* __restrict__ x,  const float* __restrict__ pos,
             const float* __restrict__ b1, const float* __restrict__ b2,
             const _Float16* __restrict__ w1f, const _Float16* __restrict__ w2f,
             float* __restrict__ out)
{
    __shared__ __attribute__((aligned(32))) _Float16 Ash[MROWS * ASTR]; // 66.0 KB
    __shared__ __attribute__((aligned(32))) _Float16 Hsh[MROWS * HSTR]; // 16.9 KB

    const int tid  = threadIdx.x;
    const int wave = tid >> 5;
    const int lane = tid & 31;
    const int half = lane >> 4;
    const int lcol = lane & 15;

    const int b   = blockIdx.x >> 5;   // 32 M-tiles per batch
    const int nb0 = (blockIdx.x & 31) * MROWS;

    v8f acc00 = {}, acc01 = {};        // m-tile 0, n-tiles 0/1
    v8f acc10 = {}, acc11 = {};        // m-tile 1, n-tiles 0/1

    for (int cc = 0; cc < TC_K1 / CHUNK; ++cc) {
        const int tb = cc * 8;         // first window-token of this K-chunk
        // ---- stage A chunk: 32 rows x 1024 f16 (x + pos), float4 reads ----
#pragma unroll
        for (int i = 0; i < 32; ++i) {
            int idx = tid * 128 + i * 4;
            int row = idx >> 10;
            int j   = idx & (CHUNK - 1);
            int t   = j >> 7;          // token within window
            int ch  = j & 127;         // channel
            int nb  = nb0 + row; if (nb > TC_NB - 1) nb = TC_NB - 1;
            int tok = nb * 16 + tb + t;
            const float4 xv = *(const float4*)(x   + ((size_t)b * TC_N + tok) * TC_C + ch);
            const float4 pv = *(const float4*)(pos + (size_t)(tb + t) * TC_C + ch);
            _Float16* d = &Ash[row * ASTR + j];
            d[0] = (_Float16)(xv.x + pv.x);
            d[1] = (_Float16)(xv.y + pv.y);
            d[2] = (_Float16)(xv.z + pv.z);
            d[3] = (_Float16)(xv.w + pv.w);
        }
        __syncthreads();

        // ---- GEMM1 over this chunk: wave owns N columns [wave*32, +32) ----
        const int nt0 = wave * 2;
        for (int kt = 0; kt < CHUNK / 32; ++kt) {
            const int kbase = kt * 32;
            // A fragments (16-bit A layout): K {0-7,16-23} / {8-15,24-31}
            const v8h a0lo = *(const v8h*)(&Ash[(lcol     ) * ASTR + kbase + half * 8]);
            const v8h a0hi = *(const v8h*)(&Ash[(lcol     ) * ASTR + kbase + 16 + half * 8]);
            const v8h a1lo = *(const v8h*)(&Ash[(lcol + 16) * ASTR + kbase + half * 8]);
            const v8h a1hi = *(const v8h*)(&Ash[(lcol + 16) * ASTR + kbase + 16 + half * 8]);
            const v16h a0 = __builtin_shufflevector(a0lo, a0hi,
                              0,1,2,3,4,5,6,7,8,9,10,11,12,13,14,15);
            const v16h a1 = __builtin_shufflevector(a1lo, a1hi,
                              0,1,2,3,4,5,6,7,8,9,10,11,12,13,14,15);
            const int ktg = cc * (CHUNK / 32) + kt;
            const v16h bf0 = *(const v16h*)(w1f + ((size_t)((nt0    ) * 128 + ktg) * 32 + lane) * 16);
            const v16h bf1 = *(const v16h*)(w1f + ((size_t)((nt0 + 1) * 128 + ktg) * 32 + lane) * 16);
            acc00 = __builtin_amdgcn_wmma_f32_16x16x32_f16(false, a0, false, bf0,
                                                           (short)0, acc00, false, false);
            acc01 = __builtin_amdgcn_wmma_f32_16x16x32_f16(false, a0, false, bf1,
                                                           (short)0, acc01, false, false);
            acc10 = __builtin_amdgcn_wmma_f32_16x16x32_f16(false, a1, false, bf0,
                                                           (short)0, acc10, false, false);
            acc11 = __builtin_amdgcn_wmma_f32_16x16x32_f16(false, a1, false, bf1,
                                                           (short)0, acc11, false, false);
        }
        __syncthreads();
    }

    // ---- bias + exact GELU, write H tile (32 x 256 f16) to LDS ----
#pragma unroll
    for (int m = 0; m < 2; ++m) {
#pragma unroll
        for (int j = 0; j < 2; ++j) {
            const v8f acc = m ? (j ? acc11 : acc10) : (j ? acc01 : acc00);
            const int col = wave * 32 + j * 16 + lcol;
            const float bias = b1[col];
#pragma unroll
            for (int v = 0; v < 8; ++v) {
                const int row = m * 16 + v + half * 8;  // D layout: M = v + 8*(lane/16)
                float val = acc[v] + bias;
                val = 0.5f * val * (1.0f + erff(val * 0.70710678118654752f));
                Hsh[row * HSTR + col] = (_Float16)val;
            }
        }
    }
    __syncthreads();

    // ---- GEMM2: M=32, N=128, K=256; wave owns columns [wave*16, +16) ----
    v8f acc2a = {}, acc2b = {};
    for (int kt = 0; kt < TC_N1 / 32; ++kt) {
        const int kbase = kt * 32;
        const v8h a0lo = *(const v8h*)(&Hsh[(lcol     ) * HSTR + kbase + half * 8]);
        const v8h a0hi = *(const v8h*)(&Hsh[(lcol     ) * HSTR + kbase + 16 + half * 8]);
        const v8h a1lo = *(const v8h*)(&Hsh[(lcol + 16) * HSTR + kbase + half * 8]);
        const v8h a1hi = *(const v8h*)(&Hsh[(lcol + 16) * HSTR + kbase + 16 + half * 8]);
        const v16h a0 = __builtin_shufflevector(a0lo, a0hi,
                          0,1,2,3,4,5,6,7,8,9,10,11,12,13,14,15);
        const v16h a1 = __builtin_shufflevector(a1lo, a1hi,
                          0,1,2,3,4,5,6,7,8,9,10,11,12,13,14,15);
        const v16h bf = *(const v16h*)(w2f + ((size_t)(wave * 8 + kt) * 32 + lane) * 16);
        acc2a = __builtin_amdgcn_wmma_f32_16x16x32_f16(false, a0, false, bf,
                                                       (short)0, acc2a, false, false);
        acc2b = __builtin_amdgcn_wmma_f32_16x16x32_f16(false, a1, false, bf,
                                                       (short)0, acc2b, false, false);
    }

    // ---- bias + fp32 store (guard the one invalid row of the last tile) ----
    const int col = wave * 16 + lcol;
    const float bias2 = b2[col];
#pragma unroll
    for (int m = 0; m < 2; ++m) {
        const v8f acc = m ? acc2b : acc2a;
#pragma unroll
        for (int v = 0; v < 8; ++v) {
            const int nb = nb0 + m * 16 + v + half * 8;
            if (nb < TC_NB)
                out[((size_t)b * TC_NB + nb) * TC_N2 + col] = acc[v] + bias2;
        }
    }
}

// ---------------------------------------------------------------------------
extern "C" void kernel_launch(void* const* d_in, const int* in_sizes, int n_in,
                              void* d_out, int out_size, void* d_ws, size_t ws_size,
                              hipStream_t stream)
{
    const float* x   = (const float*)d_in[0];
    const float* pos = (const float*)d_in[1];
    const float* W1  = (const float*)d_in[2];
    const float* b1  = (const float*)d_in[3];
    const float* W2  = (const float*)d_in[4];
    const float* b2  = (const float*)d_in[5];
    float* out = (float*)d_out;

    _Float16* w1f = (_Float16*)d_ws;                                   // 2 MB
    _Float16* w2f = (_Float16*)((char*)d_ws + (size_t)TC_K1 * TC_N1 * sizeof(_Float16));

    // W1: 16 N-tiles x 128 K-tiles; W2: 8 N-tiles x 8 K-tiles
    tc_swizzleW<<<dim3((16 * 128 * 32) / 256), dim3(256), 0, stream>>>(W1, w1f, 16, 128, TC_N1);
    tc_swizzleW<<<dim3(8), dim3(256), 0, stream>>>(W2, w2f, 8, 8, TC_N2);

    tc_main<<<dim3(TC_B * 32), dim3(256), 0, stream>>>(x, pos, b1, b2, w1f, w2f, out);
}
